// EagerSJ_56057913147482
// MI455X (gfx1250) — matrix operand order repaired and checked by reading, not compiled, not run on hardware
//
#include <hip/hip_runtime.h>

// ---------------------------------------------------------------------------
// Fused 3x3 conv (implicit GEMM, bf16 WMMA) + LIF scan for MI455X (gfx1250).
//
// Roofline: 38.7 GFLOP conv vs ~268 MB mandatory HBM traffic (~11.5us at
// 23.3 TB/s) -> fuse the LIF scan (membrane v lives in VGPRs across T) and
// run conv on v_wmma_f32_16x16x32_bf16. Weights stay register-resident
// (36 fragments/wave); activations are double-buffered in LDS and prefetched
// with global_load_async_to_lds_b128 (ASYNCcnt) one time-step ahead.
// sched_group_barrier pins a depth-2 A-fragment pipeline: DS reads run two
// fragments ahead of the consuming WMMAs (s_wait_dscnt <= 4, not 0).
//
// Block = (n, h) row: 8 waves. wave&3 -> 16-pixel w-tile, wave>>2 -> 32-cout
// half. ~400 VGPRs/wave -> 2 waves/SIMD.
// ---------------------------------------------------------------------------

typedef __attribute__((ext_vector_type(16))) __bf16 v16bf;
typedef __attribute__((ext_vector_type(8)))  float  v8f;

#define T_STEPS 16
#define NBATCH  8
#define CH      64
#define HH      64
#define WW      64
#define ROWU16  (66 * CH)            // one staged halo row, u16 elements
#define BUFU16  (3 * ROWU16)         // one t-step buffer (3 rows), u16 elements

#if defined(__has_builtin)
#if __has_builtin(__builtin_amdgcn_sched_group_barrier)
#define SCHED_GROUP(mask, size, id) __builtin_amdgcn_sched_group_barrier((mask), (size), (id))
#endif
#endif
#ifndef SCHED_GROUP
#define SCHED_GROUP(mask, size, id)
#endif

static __device__ __forceinline__ unsigned short f2bf(float f) {
    unsigned u = __float_as_uint(f);
    unsigned r = ((u >> 16) & 1u) + 0x7FFFu;      // round to nearest even
    return (unsigned short)((u + r) >> 16);
}

// --------------------------------------------------------------------------
// Prep 1: weights -> WMMA B-layout bf16 fragments (32Kx16N each).
// B k order used here (contiguous, not AMD's table order): lanes 0-15 hold K=0..15, lanes 16-31 hold K=16..31;
// per-lane fragment = 16 contiguous bf16 (32 B).
// --------------------------------------------------------------------------
__global__ void prep_weights_kernel(const float* __restrict__ w,
                                    unsigned short* __restrict__ wsw) {
    int i = blockIdx.x * 256 + threadIdx.x;
    if (i >= 9 * 2 * 4 * 32 * 16) return;
    int e    = i & 15;
    int lane = (i >> 4) & 31;
    int nt   = (i >> 9) & 3;
    int c    = (i >> 11) & 1;
    int tap  = i >> 12;
    int co   = nt * 16 + (lane & 15);
    int cin  = c * 32 + (lane >> 4) * 16 + e;
    wsw[i] = f2bf(w[(co * 64 + cin) * 9 + tap]);
}

// --------------------------------------------------------------------------
// Prep 2: x fp32 NCHW -> bf16 NHWC with channels in natural order
// so each lane's A-fragment (lanes 0-15 hold K 0..15, lanes 16-31 hold
// K 16..31, matching B) is one 32 B read.
// --------------------------------------------------------------------------
__global__ void prep_input_kernel(const float* __restrict__ x,
                                  unsigned short* __restrict__ xs, int total) {
    int i = blockIdx.x * 256 + threadIdx.x;
    if (i >= total) return;
    int w_ = i & 63;
    int h_ = (i >> 6) & 63;
    int ch = (i >> 12) & 63;
    int tn = i >> 18;
    int pos = ch;
    size_t o = (((size_t)tn * (HH * WW)) + h_ * WW + w_) * CH + pos;
    xs[o] = f2bf(x[i]);
}

// --------------------------------------------------------------------------
// Main fused kernel.
// --------------------------------------------------------------------------
__global__ __launch_bounds__(256)
void lif_conv_main(const unsigned short* __restrict__ xs,
                   const unsigned short* __restrict__ wsw,
                   float* __restrict__ out) {
    __shared__ __attribute__((aligned(16))) unsigned short lds[2 * BUFU16];

    const int tid   = threadIdx.x;
    const int lane  = tid & 31;
    const int wave  = tid >> 5;
    const int wtile = wave & 3;       // w-tile 0..3
    const int nth   = wave >> 2;      // cout half 0..1
    const int n     = blockIdx.x >> 6;
    const int h     = blockIdx.x & 63;
    const int w0    = wtile * 16;
    const int m     = lane & 15;      // A-matrix row (pixel)
    const int half  = lane >> 4;
    const bool edge = (h == 0) || (h == 63);
    const unsigned ldsBase = (unsigned)(size_t)(&lds[0]);

    // ---- preload 36 B fragments (t-invariant) into VGPRs -----------------
    v16bf bfrag[36];
    #pragma unroll
    for (int f = 0; f < 36; ++f) {
        const int tap = f >> 2;
        const int c   = (f >> 1) & 1;
        const int ntl = f & 1;
        const int gfrag = (tap * 2 + c) * 4 + (nth * 2 + ntl);
        bfrag[f] = *reinterpret_cast<const v16bf*>(wsw + (gfrag * 32 + lane) * 16);
    }

    // ---- zero both LDS buffers once (halo cells stay zero forever) -------
    for (int i = tid; i < (2 * BUFU16 * 2) / 16; i += 256)
        reinterpret_cast<uint4*>(lds)[i] = make_uint4(0u, 0u, 0u, 0u);
    __syncthreads();

    // async-stage the 2-3 valid halo rows of time-step tt into buffer tt&1
    auto issue_rows = [&](int tt) {
        const size_t imgB = (size_t)(tt * NBATCH + n) * (HH * WW * CH) * 2;
        const unsigned bufB = ldsBase + (unsigned)((tt & 1) * BUFU16 * 2);
        const int r0 = (h == 0) ? 1 : 0;
        const int r1 = (h == 63) ? 1 : 2;
        const int nch = (r1 - r0 + 1) * 512;         // 16B chunks (uniform)
        for (int i = tid; i < nch; i += 256) {
            const int r    = r0 + (i >> 9);
            const int rem  = i & 511;
            const int px   = (rem >> 3) + 1;          // interior 1..64
            const int part = rem & 7;
            const int gh   = h + r - 1;
            unsigned long long g = (unsigned long long)(size_t)xs + imgB +
                ((size_t)(gh * WW + (px - 1)) * CH + (size_t)part * 8) * 2;
            unsigned l = bufB + (unsigned)(((r * 66 + px) * CH + part * 8) * 2);
            asm volatile("global_load_async_to_lds_b128 %0, %1, off"
                         :: "v"(l), "v"(g) : "memory");
        }
    };

    issue_rows(0);                    // 6 (or 4) async ops per wave

    v8f vstate[2] = {};               // membrane potential, V_RESET = 0

    #pragma unroll 1
    for (int t = 0; t < T_STEPS; ++t) {
        __syncthreads();              // all waves done reading buf[(t-1)&1]
        if (t + 1 < T_STEPS) issue_rows(t + 1);      // prefetch into buf[(t+1)&1]

        // Wait until only the just-issued copies remain outstanding ->
        // this step's copies (issued last iteration) have landed.
        if (t + 1 >= T_STEPS) {
            asm volatile("s_wait_asynccnt 0x0" ::: "memory");
        } else if (edge) {
            asm volatile("s_wait_asynccnt 0x4" ::: "memory");
        } else {
            asm volatile("s_wait_asynccnt 0x6" ::: "memory");
        }
        __syncthreads();              // everyone's copies for step t visible

        const int bufOff = (t & 1) * BUFU16;

        // A-fragment LDS offset for K-step k (tap = k>>1, chunk = k&1)
        auto aoffset = [&](int k) -> int {
            const int tap = k >> 1;
            const int c   = k & 1;
            const int dy  = tap / 3 - 1;
            const int dx  = tap % 3 - 1;
            return bufOff + ((1 + dy) * 66 + (w0 + m + dx + 1)) * CH
                   + c * 32 + half * 16;
        };
        auto loadA = [&](int k) -> v16bf {
            return *reinterpret_cast<const v16bf*>(&lds[aoffset(k)]);
        };

        // ---- 18 K-steps, depth-2 software-pipelined A loads --------------
        v8f acc[2] = {};
        v16bf a0 = loadA(0);
        v16bf a1 = loadA(1);
        #pragma unroll
        for (int k = 0; k < 18; ++k) {
            v16bf an;
            if (k + 2 < 18) an = loadA(k + 2);
            #pragma unroll
            for (int ntl = 0; ntl < 2; ++ntl) {
                acc[ntl] = __builtin_amdgcn_wmma_f32_16x16x32_bf16(
                    false, a0, false, bfrag[k * 2 + ntl],
                    (short)0, acc[ntl], false, false);
            }
            a0 = a1;
            a1 = an;
        }

        // Pin the pipeline: DS reads (0x100) run 2 fragments ahead of the
        // consuming WMMAs (0x008) -> s_wait_dscnt <= 4 instead of 0.
        SCHED_GROUP(0x100, 4, 0);                 // frags 0,1 (4 ds reads)
        #pragma unroll
        for (int k = 0; k < 16; ++k) {
            SCHED_GROUP(0x008, 2, 0);             // WMMA pair for step k
            SCHED_GROUP(0x100, 2, 0);             // ds reads for frag k+2
        }
        SCHED_GROUP(0x008, 4, 0);                 // WMMA pairs for steps 16,17

        // LIF update + spike store. D layout: VGPR r -> pixel M = r + half*8,
        // lane%16 -> cout within 16-wide ntile; 8 consecutive w per lane.
        #pragma unroll
        for (int ntl = 0; ntl < 2; ++ntl) {
            const int cout = (nth * 2 + ntl) * 16 + m;
            float s[8];
            #pragma unroll
            for (int r = 0; r < 8; ++r) {
                float z  = acc[ntl][r];
                float vv = vstate[ntl][r] + (z - vstate[ntl][r]) * 0.5f; // tau=2
                bool  sp = (vv >= 1.0f);
                s[r] = sp ? 1.0f : 0.0f;
                vstate[ntl][r] = sp ? 0.0f : vv;     // hard reset to 0
            }
            float* dst = out + ((size_t)(t * NBATCH + n) * CH + cout) * (HH * WW)
                         + h * WW + w0 + half * 8;
            *reinterpret_cast<float4*>(dst)     = make_float4(s[0], s[1], s[2], s[3]);
            *reinterpret_cast<float4*>(dst + 4) = make_float4(s[4], s[5], s[6], s[7]);
        }
    }
}

extern "C" void kernel_launch(void* const* d_in, const int* in_sizes, int n_in,
                              void* d_out, int out_size, void* d_ws, size_t ws_size,
                              hipStream_t stream) {
    const float* x = (const float*)d_in[0];   // (16,8,64,64,64) fp32
    const float* w = (const float*)d_in[1];   // (64,64,3,3) fp32
    float* out = (float*)d_out;

    unsigned short* wsw = (unsigned short*)d_ws;                       // 73,728 B
    unsigned short* xs  = (unsigned short*)((char*)d_ws + (1 << 17));  // +128 KB

    (void)in_sizes; (void)n_in; (void)out_size; (void)ws_size;

    prep_weights_kernel<<<(9 * 2 * 4 * 32 * 16 + 255) / 256, 256, 0, stream>>>(w, wsw);

    const int total = T_STEPS * NBATCH * CH * HH * WW;   // 33,554,432
    prep_input_kernel<<<total / 256, 256, 0, stream>>>(x, xs, total);

    lif_conv_main<<<NBATCH * HH, 256, 0, stream>>>(xs, wsw, out);
}
